// FP8Linear_39548058861700
// MI455X (gfx1250) — 8-bit matrix operand order repaired and checked by reading, compiled, not run on hardware
//
#include <hip/hip_runtime.h>


typedef __attribute__((ext_vector_type(16))) int   v16i;
typedef __attribute__((ext_vector_type(8)))  float v8f;

#define MDIM 8192
#define NDIM 8192
#define KDIM 2048
#define BM   128
#define BN   128
#define BK   128
#define NT   (KDIM / BK)
#define LDT  144

__device__ __forceinline__ unsigned char f32_to_e4m3(float x) {
  unsigned int u = __float_as_uint(x);
  unsigned int s = (u >> 24) & 0x80u;
  float a = __builtin_fabsf(x);
  if (a != a) return (unsigned char)(s | 0x7Fu);
  if (a >= 464.0f) return (unsigned char)(s | 0x7Eu);
  unsigned int au = __float_as_uint(a);
  int E = (int)(au >> 23) - 127;
  if (E < -6) {
    int m = (int)__builtin_rintf(a * 512.0f);
    return (unsigned char)(s | (unsigned int)m);
  }
  unsigned int mant = au & 0x7FFFFFu;
  unsigned int m3   = mant >> 20;
  unsigned int rest = mant & 0xFFFFFu;
  if (rest > 0x80000u || (rest == 0x80000u && (m3 & 1u))) m3++;
  unsigned int e4 = (unsigned int)(E + 7);
  if (m3 == 8u) { m3 = 0u; e4++; }
  if (e4 > 15u || (e4 == 15u && m3 == 7u)) return (unsigned char)(s | 0x7Eu);
  return (unsigned char)(s | (e4 << 3) | m3);
}

__global__ __launch_bounds__(256)
void quant_e4m3_kernel(const float* __restrict__ x,
                       const float* __restrict__ scale,
                       unsigned char* __restrict__ q, int n4) {
  int i = blockIdx.x * blockDim.x + threadIdx.x;
  if (i >= n4) return;
  float s = scale[0];
  float4 v = ((const float4*)x)[i];
  uchar4 r;
  r.x = f32_to_e4m3(v.x * s);
  r.y = f32_to_e4m3(v.y * s);
  r.z = f32_to_e4m3(v.z * s);
  r.w = f32_to_e4m3(v.w * s);
  ((uchar4*)q)[i] = r;
}

__device__ __forceinline__ void async_ld_b128(unsigned lds_addr, const void* g) {
  unsigned long long ga = (unsigned long long)(uintptr_t)g;
  asm volatile("global_load_async_to_lds_b128 %0, %1, off"
               :: "v"(lds_addr), "v"(ga) : "memory");
}
__device__ __forceinline__ void wait_async0() {
  asm volatile("s_wait_asynccnt 0x0" ::: "memory");
}

__global__ __launch_bounds__(256)
void fp8_gemm_wmma_kernel(const unsigned char* __restrict__ Aq,
                          const unsigned char* __restrict__ Bq,
                          const float* __restrict__ sx,
                          const float* __restrict__ sw,
                          float* __restrict__ out) {
  __shared__ unsigned char lds_a[2][BM * LDT];
  __shared__ unsigned char lds_b[2][BN * LDT];

  const int tid  = threadIdx.x;
  const int lane = tid & 31;
  const int wave = tid >> 5;

  const int mTile = blockIdx.y * BM;
  const int nTile = blockIdx.x * BN;

  const int mW = (wave & 3) * 32;
  const int nW = (wave >> 2) * 64;

  const int half = lane >> 4;
  const int l16  = lane & 15;

  const int srow = tid >> 3;
  const int scol = (tid & 7) * 16;

  const unsigned ldsA[2] = { (unsigned)(uintptr_t)&lds_a[0][0],
                             (unsigned)(uintptr_t)&lds_a[1][0] };
  const unsigned ldsB[2] = { (unsigned)(uintptr_t)&lds_b[0][0],
                             (unsigned)(uintptr_t)&lds_b[1][0] };

  v8f acc[2][4];
  {
    v8f z = {0.f, 0.f, 0.f, 0.f, 0.f, 0.f, 0.f, 0.f};
    #pragma unroll
    for (int mi = 0; mi < 2; ++mi)
      #pragma unroll
      for (int ni = 0; ni < 4; ++ni) acc[mi][ni] = z;
  }

  auto issue_tile = [&](int kt, int buf) {
    const int k0 = kt * BK;
    #pragma unroll
    for (int r = 0; r < 4; ++r) {
      const int row = srow + 32 * r;
      async_ld_b128(ldsA[buf] + row * LDT + scol,
                    Aq + (size_t)(mTile + row) * KDIM + k0 + scol);
      async_ld_b128(ldsB[buf] + row * LDT + scol,
                    Bq + (size_t)(nTile + row) * KDIM + k0 + scol);
    }
  };

  issue_tile(0, 0);
  wait_async0();
  __syncthreads();

  for (int kt = 0; kt < NT; ++kt) {
    const int cur = kt & 1;

    if (kt + 1 < NT) issue_tile(kt + 1, cur ^ 1);

    v16i bf[4];
    #pragma unroll
    for (int ni = 0; ni < 4; ++ni) {
      const int bo = (nW + 16 * ni + l16) * LDT + 16 * half;
      #pragma unroll
      for (int j = 0; j < 4; ++j) {
        int4 d = *(const int4*)&lds_b[cur][bo + 32 * j];
        bf[ni][4 * j + 0] = d.x; bf[ni][4 * j + 1] = d.y;
        bf[ni][4 * j + 2] = d.z; bf[ni][4 * j + 3] = d.w;
      }
    }

    #pragma unroll
    for (int mi = 0; mi < 2; ++mi) {
      v16i af;
      const int ao = (mW + 16 * mi + l16) * LDT + 16 * half;
      #pragma unroll
      for (int j = 0; j < 4; ++j) {
        int4 d = *(const int4*)&lds_a[cur][ao + 32 * j];
        af[4 * j + 0] = d.x; af[4 * j + 1] = d.y;
        af[4 * j + 2] = d.z; af[4 * j + 3] = d.w;
      }
      #pragma unroll
      for (int ni = 0; ni < 4; ++ni) {
        acc[mi][ni] = __builtin_amdgcn_wmma_f32_16x16x128_fp8_fp8(
            af, bf[ni], (short)0, acc[mi][ni],
             false,  false);
      }
    }

    wait_async0();
    __syncthreads();
  }

  const float inv = 1.0f / (sx[0] * sw[0]);
  #pragma unroll
  for (int mi = 0; mi < 2; ++mi) {
    #pragma unroll
    for (int ni = 0; ni < 4; ++ni) {
      const int n = nTile + nW + 16 * ni + l16;
      #pragma unroll
      for (int i = 0; i < 8; ++i) {
        const int m = mTile + mW + 16 * mi + 8 * half + i;
        out[(size_t)m * NDIM + n] = acc[mi][ni][i] * inv;
      }
    }
  }
}

extern "C" void kernel_launch(void* const* d_in, const int* in_sizes, int n_in,
                              void* d_out, int out_size, void* d_ws, size_t ws_size,
                              hipStream_t stream) {
  (void)in_sizes; (void)n_in; (void)out_size; (void)ws_size;
  const float* x  = (const float*)d_in[0];
  const float* w  = (const float*)d_in[1];
  const float* sx = (const float*)d_in[2];
  const float* sw = (const float*)d_in[3];

  unsigned char* Aq = (unsigned char*)d_ws;
  unsigned char* Wq = Aq + (size_t)MDIM * KDIM;

  const int n4 = (MDIM * KDIM) / 4;
  quant_e4m3_kernel<<<n4 / 256, 256, 0, stream>>>(x, sx, Aq, n4);
  quant_e4m3_kernel<<<n4 / 256, 256, 0, stream>>>(w, sw, Wq, n4);

  dim3 grid(NDIM / BN, MDIM / BM);
  fp8_gemm_wmma_kernel<<<grid, 256, 0, stream>>>(Aq, Wq, sx, sw, (float*)d_out);
}
